// SO2Net_30983894073419
// MI455X (gfx1250) — compile-verified, matrix operand order checked by reading, not run on hardware
//
#include <hip/hip_runtime.h>
#include <hip/hip_bf16.h>
#include <stdint.h>

#define NCOEF    25
#define C_DIM    128
#define NG_DIM   128
#define N_NODES  2000
#define N_EDGES  32000
#define D_DIM    3200           // NCOEF * C_DIM
#define CUTOFF_F 13.0f

typedef __attribute__((ext_vector_type(16))) _Float16 v16h;
typedef __attribute__((ext_vector_type(8)))  _Float16 v8h;
typedef __attribute__((ext_vector_type(8)))  float    v8f;

__device__ __forceinline__ float silu_f(float x) {
    return x / (1.0f + __expf(-x));
}

// Low 32 bits of a generic pointer to LDS == LDS offset on AMDGPU.
__device__ __forceinline__ unsigned lds_u32(const void* p) {
    return (unsigned)(uintptr_t)p;
}

// ---------------------------------------------------------------------------
// Gaussian smearing: dist_emb[e][g] = exp(coeff * (d_e - offset_g)^2)
// ---------------------------------------------------------------------------
__global__ void k_smear(const float* __restrict__ edge_attr,
                        float* __restrict__ dist_emb) {
    int e = blockIdx.x;
    int g = threadIdx.x;               // 128 threads
    float d    = edge_attr[e * 4 + 0];
    float step = CUTOFF_F / (float)(NG_DIM - 1);
    float off  = (float)g * step;
    float coeff = -0.5f / ((2.0f * step) * (2.0f * step));
    float x = d - off;
    dist_emb[(size_t)e * NG_DIM + g] = __expf(coeff * x * x);
}

// ---------------------------------------------------------------------------
// Per-edge rotation matrix (deterministic hash stands in for JAX PRNG)
// ---------------------------------------------------------------------------
__device__ __forceinline__ float hash_u01(uint32_t x) {
    x ^= x >> 17; x *= 0xed5ad4bbu;
    x ^= x >> 11; x *= 0xac4c1b51u;
    x ^= x >> 15; x *= 0x31848babu;
    x ^= x >> 14;
    return (float)(x >> 8) * (1.0f / 16777216.0f);
}

__global__ void k_rotmat(const float* __restrict__ edge_attr,
                         float* __restrict__ R) {
    int e = blockIdx.x * blockDim.x + threadIdx.x;
    if (e >= N_EDGES) return;
    // vec = edge_attr[:, (2,3,1)]
    float vx = edge_attr[e * 4 + 2];
    float vy = edge_attr[e * 4 + 3];
    float vz = edge_attr[e * 4 + 1];
    float d = sqrtf(vx * vx + vy * vy + vz * vz) + 1e-12f;
    float nx0 = vx / d, nx1 = vy / d, nx2 = vz / d;

    float a0 = hash_u01((uint32_t)(3 * e + 1)) - 0.5f;
    float a1 = hash_u01((uint32_t)(3 * e + 2)) - 0.5f;
    float a2 = hash_u01((uint32_t)(3 * e + 3)) - 0.5f;
    float n = sqrtf(a0 * a0 + a1 * a1 + a2 * a2) + 1e-12f;
    float v20 = a0 / n, v21 = a1 / n, v22 = a2 / n;

    float b0 = -v21, b1 = v20,  b2 = v22;
    float c0 = v20,  c1 = -v22, c2 = v21;
    float dv = fabsf(v20 * nx0 + v21 * nx1 + v22 * nx2);
    float db = fabsf(b0 * nx0 + b1 * nx1 + b2 * nx2);
    if (dv > db) { v20 = b0; v21 = b1; v22 = b2; dv = db; }
    float dc = fabsf(c0 * nx0 + c1 * nx1 + c2 * nx2);
    if (dv > dc) { v20 = c0; v21 = c1; v22 = c2; }

    float z0 = nx1 * v22 - nx2 * v21;
    float z1 = nx2 * v20 - nx0 * v22;
    float z2 = nx0 * v21 - nx1 * v20;
    float zn = sqrtf(z0 * z0 + z1 * z1 + z2 * z2) + 1e-12f;
    z0 /= zn; z1 /= zn; z2 /= zn;

    float y0 = nx1 * z2 - nx2 * z1;
    float y1 = nx2 * z0 - nx0 * z2;
    float y2 = nx0 * z1 - nx1 * z0;
    float yn = sqrtf(y0 * y0 + y1 * y1 + y2 * y2) + 1e-12f;
    y0 /= yn; y1 /= yn; y2 /= yn;

    // R rows: nz, nx, -ny
    R[e * 9 + 0] = z0;  R[e * 9 + 1] = z1;  R[e * 9 + 2] = z2;
    R[e * 9 + 3] = nx0; R[e * 9 + 4] = nx1; R[e * 9 + 5] = nx2;
    R[e * 9 + 6] = -y0; R[e * 9 + 7] = -y1; R[e * 9 + 8] = -y2;
}

// ---------------------------------------------------------------------------
// Embedding init
// ---------------------------------------------------------------------------
__global__ void k_init_node(const int* __restrict__ atomic_numbers,
                            const float* __restrict__ sphere_emb,
                            float* __restrict__ node_emb) {
    unsigned idx = blockIdx.x * blockDim.x + threadIdx.x;
    if (idx >= (unsigned)(N_NODES * D_DIM)) return;
    unsigned nrow = idx / D_DIM;
    unsigned r    = idx % D_DIM;
    float v = 0.0f;
    if (r < C_DIM) v = sphere_emb[atomic_numbers[nrow] * C_DIM + r];
    node_emb[idx] = v;
}

__global__ void k_init_edge(const float* __restrict__ dist_emb,
                            float* __restrict__ edge_emb) {
    unsigned idx = blockIdx.x * blockDim.x + threadIdx.x;
    if (idx >= (unsigned)(N_EDGES * D_DIM)) return;
    unsigned e = idx / D_DIM;
    unsigned r = idx % D_DIM;
    float v = 0.0f;
    if (r < C_DIM) v = dist_emb[(size_t)e * NG_DIM + r];
    edge_emb[idx] = v;
}

// ---------------------------------------------------------------------------
// Generic WMMA GEMM: Out(MxN) (+)= silu?( A(MxK) @ B(KxN) + bias? )
// f32 in/out, f16 WMMA accumulate-in-f32. K % 32 == 0, N % 128 == 0.
// flags: 1 = accumulate into Out, 2 = SiLU epilogue, 4 = add bias[col]
// Block: 256 threads (8 wave32); tile 128x128x32; wave tile 64x32 (4x2 WMMA).
// Software pipeline: double-buffered GLOBAL_LOAD_ASYNC_TO_LDS_B128 staging —
// tile k+1 DMA overlaps the WMMA compute of tile k. Per-thread global
// addresses are precomputed and bumped by constant byte strides.
// ---------------------------------------------------------------------------
#define BM 128
#define BN 128
#define BK 32

#define ASTAGE_BYTES (BM * BK * 4)     // 16 KB per buffer
#define BSTAGE_BYTES (BK * BN * 4)     // 16 KB per buffer

__global__ __launch_bounds__(256)
void k_gemm_wmma(const float* __restrict__ A, const float* __restrict__ B,
                 const float* __restrict__ bias, float* __restrict__ Out,
                 int M, int N, int K, int flags) {
    __shared__ __align__(16) float    Astage[2][BM][BK];    // 32 KB (async dst)
    __shared__ __align__(16) float    Bstage[2][BK][BN];    // 32 KB (async dst)
    __shared__ __align__(16) _Float16 As[BM][BK + 8];       // 10 KB, row-major K
    __shared__ __align__(16) _Float16 Bs[BN][BK + 8];       // 10 KB, N-major!

    const int tid   = threadIdx.x;
    const int lane  = tid & 31;
    const int wid   = tid >> 5;        // 0..7
    const int wm    = wid & 1;         // row half
    const int wn    = wid >> 1;        // 0..3 column quarter
    const int lhalf = lane >> 4;       // 0/1
    const int lrow  = lane & 15;
    const int kbase = lhalf << 3;      // 0 or 8

    const int m0 = blockIdx.x * BM;
    const int n0 = blockIdx.y * BN;

    // ---- precompute per-thread async-copy addresses (byte-incremental) ----
    unsigned long long gaA[4], gaB[4];
    unsigned laA[4], laB[4];
#pragma unroll
    for (int i = 0; i < 4; ++i) {
        int id = i * 256 + tid;               // A: 1024 16B chunks of 128x32
        int r  = id >> 3;                     // 0..127
        int cq = id & 7;                      // float4 column
        int gr = m0 + r; if (gr >= M) gr = M - 1;          // branchless guard
        gaA[i] = (unsigned long long)(uintptr_t)(A + (size_t)gr * K + cq * 4);
        laA[i] = lds_u32(&Astage[0][r][cq * 4]);
    }
#pragma unroll
    for (int i = 0; i < 4; ++i) {
        int id = i * 256 + tid;               // B: 1024 16B chunks of 32x128
        int br = id >> 5;                     // 0..31
        int bc = id & 31;                     // float4 column
        gaB[i] = (unsigned long long)(uintptr_t)(B + (size_t)br * N + n0 + bc * 4);
        laB[i] = lds_u32(&Bstage[0][br][bc * 4]);
    }
    const unsigned long long incA = (unsigned long long)BK * 4;            // bytes
    const unsigned long long incB = (unsigned long long)BK * N * 4;        // bytes

    const v8f vzero = {0.f, 0.f, 0.f, 0.f, 0.f, 0.f, 0.f, 0.f};
    v8f acc[4][2];
#pragma unroll
    for (int i = 0; i < 4; ++i)
#pragma unroll
        for (int j = 0; j < 2; ++j) acc[i][j] = vzero;

    // ---- prologue: DMA tile 0 into stage buffer 0 ----
#pragma unroll
    for (int i = 0; i < 4; ++i) {
        asm volatile("global_load_async_to_lds_b128 %0, %1, off"
                     :: "v"(laA[i]), "v"(gaA[i]) : "memory");
        gaA[i] += incA;
    }
#pragma unroll
    for (int i = 0; i < 4; ++i) {
        asm volatile("global_load_async_to_lds_b128 %0, %1, off"
                     :: "v"(laB[i]), "v"(gaB[i]) : "memory");
        gaB[i] += incB;
    }

    int p = 0;
    for (int k0 = 0; k0 < K; k0 += BK) {
        asm volatile("s_wait_asynccnt 0x0" ::: "memory");
        __syncthreads();   // (a) stage[p] visible; prior compute finished

        // ---- convert f32 stage[p] -> f16 fragment-contiguous layouts ----
#pragma unroll
        for (int i = 0; i < 2; ++i) {         // A: 512 row-octet tasks
            int t2 = i * 256 + tid;
            int r  = t2 >> 2;                 // 0..127
            int o  = t2 & 3;                  // k-octet
            float fm = ((m0 + r) < M) ? 1.0f : 0.0f;
            float4 fa = *(const float4*)&Astage[p][r][o * 8];
            float4 fb = *(const float4*)&Astage[p][r][o * 8 + 4];
            v8h h;
            h[0] = (_Float16)(fa.x * fm); h[1] = (_Float16)(fa.y * fm);
            h[2] = (_Float16)(fa.z * fm); h[3] = (_Float16)(fa.w * fm);
            h[4] = (_Float16)(fb.x * fm); h[5] = (_Float16)(fb.y * fm);
            h[6] = (_Float16)(fb.z * fm); h[7] = (_Float16)(fb.w * fm);
            *(v8h*)&As[r][o * 8] = h;
        }
#pragma unroll
        for (int i = 0; i < 2; ++i) {         // B: 512 col-octet tasks (transpose)
            int t2 = i * 256 + tid;
            int c  = t2 >> 2;                 // 0..127
            int o  = t2 & 3;
            v8h h;
#pragma unroll
            for (int j = 0; j < 8; ++j)
                h[j] = (_Float16)Bstage[p][o * 8 + j][c];
            *(v8h*)&Bs[c][o * 8] = h;
        }
        __syncthreads();   // (b) fragments ready; stage[p^1] free for DMA

        // ---- issue DMA for tile k+1 into the other stage (overlaps WMMA) ----
        if (k0 + BK < K) {
            unsigned offA = (p ^ 1) ? (unsigned)ASTAGE_BYTES : 0u;
            unsigned offB = (p ^ 1) ? (unsigned)BSTAGE_BYTES : 0u;
#pragma unroll
            for (int i = 0; i < 4; ++i) {
                asm volatile("global_load_async_to_lds_b128 %0, %1, off"
                             :: "v"(laA[i] + offA), "v"(gaA[i]) : "memory");
                gaA[i] += incA;
            }
#pragma unroll
            for (int i = 0; i < 4; ++i) {
                asm volatile("global_load_async_to_lds_b128 %0, %1, off"
                             :: "v"(laB[i] + offB), "v"(gaB[i]) : "memory");
                gaB[i] += incB;
            }
            // L2 deep-prefetch two tiles ahead (global_prefetch_b8)
            if (k0 + 2 * BK < K)
                __builtin_prefetch(&B[(size_t)(k0 + 2 * BK + (tid >> 5)) * N + n0 + ((tid & 31) << 2)], 0, 1);
        }

        // ---- fragments: two contiguous 16B LDS chunks each ----
        v16h afrag[4];
#pragma unroll
        for (int mt = 0; mt < 4; ++mt) {
            int r = wm * 64 + mt * 16 + lrow;
            v8h lo = *(const v8h*)&As[r][kbase];
            v8h hi = *(const v8h*)&As[r][kbase + 16];
            afrag[mt] = __builtin_shufflevector(lo, hi, 0, 1, 2, 3, 4, 5, 6, 7,
                                                8, 9, 10, 11, 12, 13, 14, 15);
        }
        v16h bfrag[2];
#pragma unroll
        for (int nt = 0; nt < 2; ++nt) {
            int c = wn * 32 + nt * 16 + lrow;
            v8h lo = *(const v8h*)&Bs[c][kbase];
            v8h hi = *(const v8h*)&Bs[c][kbase + 16];
            bfrag[nt] = __builtin_shufflevector(lo, hi, 0, 1, 2, 3, 4, 5, 6, 7,
                                                8, 9, 10, 11, 12, 13, 14, 15);
        }

#pragma unroll
        for (int mt = 0; mt < 4; ++mt)
#pragma unroll
            for (int nt = 0; nt < 2; ++nt)
                acc[mt][nt] = __builtin_amdgcn_wmma_f32_16x16x32_f16(
                    false, afrag[mt], false, bfrag[nt],
                    (short)0, acc[mt][nt], false, false);

        p ^= 1;
    }

    // ---- epilogue ----
#pragma unroll
    for (int mt = 0; mt < 4; ++mt) {
#pragma unroll
        for (int nt = 0; nt < 2; ++nt) {
            int col = n0 + wn * 32 + nt * 16 + lrow;
#pragma unroll
            for (int v = 0; v < 8; ++v) {
                int row = m0 + wm * 64 + mt * 16 + lhalf * 8 + v;
                if (row < M) {
                    float val = acc[mt][nt][v];
                    if (flags & 4) val += bias[col];
                    if (flags & 2) val = silu_f(val);
                    size_t o = (size_t)row * N + col;
                    if (flags & 1) Out[o] += val;
                    else           Out[o] = val;
                }
            }
        }
    }
}

// ---------------------------------------------------------------------------
// Message: m = silu(node[src] + node[tgt] + edge) * rad ; rotate l=1 block
// transposeR == 0: out_i = sum_j R[i][j] m_j ; == 1: out_i = sum_j R[j][i] m_j
// ---------------------------------------------------------------------------
__global__ __launch_bounds__(256)
void k_message(const float* __restrict__ node_emb, const float* __restrict__ edge_emb,
               const float* __restrict__ rad, const int* __restrict__ src,
               const int* __restrict__ tgt, const float* __restrict__ R,
               float* __restrict__ msg, int transposeR) {
    __shared__ float buf[3][C_DIM];
    int e = blockIdx.x;
    int s = src[e], t = tgt[e];
    const float* ns  = node_emb + (size_t)s * D_DIM;
    const float* nt  = node_emb + (size_t)t * D_DIM;
    const float* ee  = edge_emb + (size_t)e * D_DIM;
    const float* rde = rad + (size_t)e * NG_DIM;
    float* me = msg + (size_t)e * D_DIM;

    for (int idx = threadIdx.x; idx < D_DIM; idx += 256) {
        int l = idx >> 7, c = idx & 127;
        float x = silu_f(ns[idx] + nt[idx] + ee[idx]) * rde[c];
        if (l >= 1 && l <= 3) buf[l - 1][c] = x;
        else                  me[idx] = x;
    }
    __syncthreads();
    if (threadIdx.x < C_DIM) {
        int c = threadIdx.x;
        float r[9];
#pragma unroll
        for (int i = 0; i < 9; ++i) r[i] = R[e * 9 + i];
#pragma unroll
        for (int i = 0; i < 3; ++i) {
            float a = 0.0f;
#pragma unroll
            for (int j = 0; j < 3; ++j) {
                float w = transposeR ? r[j * 3 + i] : r[i * 3 + j];
                a += w * buf[j][c];
            }
            me[(1 + i) * C_DIM + c] = a;
        }
    }
}

// ---------------------------------------------------------------------------
// Zero fill + segment-sum scatter (global_atomic_add_f32)
// ---------------------------------------------------------------------------
__global__ void k_zero(float* __restrict__ p, unsigned n) {
    unsigned i = blockIdx.x * blockDim.x + threadIdx.x;
    if (i < n) p[i] = 0.0f;
}

__global__ void k_scatter(const float* __restrict__ msg, const int* __restrict__ tgt,
                          float* __restrict__ agg) {
    unsigned idx = blockIdx.x * blockDim.x + threadIdx.x;
    if (idx >= (unsigned)(N_EDGES * D_DIM)) return;
    unsigned e = idx / D_DIM;
    unsigned r = idx % D_DIM;
    atomicAdd(&agg[(size_t)tgt[e] * D_DIM + r], msg[idx]);
}

// ---------------------------------------------------------------------------
// irreps reorder: X[n][base_l + c*(2l+1) + m] = emb[n][l*l + m][c]
// ---------------------------------------------------------------------------
__global__ void k_irreps(const float* __restrict__ emb, float* __restrict__ X,
                         int n_rows) {
    unsigned idx = blockIdx.x * blockDim.x + threadIdx.x;
    if (idx >= (unsigned)n_rows * (unsigned)D_DIM) return;
    unsigned nrow = idx / D_DIM;
    unsigned f    = idx % D_DIM;
    int l, base;
    if      (f < 128)  { l = 0; base = 0;    }
    else if (f < 512)  { l = 1; base = 128;  }
    else if (f < 1152) { l = 2; base = 512;  }
    else if (f < 2048) { l = 3; base = 1152; }
    else               { l = 4; base = 2048; }
    int q  = (int)f - base;
    int w  = 2 * l + 1;
    int c  = q / w;
    int mm = q % w;
    int coef = l * l + mm;
    X[idx] = emb[(size_t)nrow * D_DIM + coef * C_DIM + c];
}

// ---------------------------------------------------------------------------
extern "C" void kernel_launch(void* const* d_in, const int* in_sizes, int n_in,
                              void* d_out, int out_size, void* d_ws, size_t ws_size,
                              hipStream_t stream) {
    const int*   atomic_numbers = (const int*)d_in[0];
    const float* edge_attr      = (const float*)d_in[1];
    const int*   edge_index     = (const int*)d_in[2];
    const float* sphere_emb     = (const float*)d_in[3];
    const float* W_rad_n        = (const float*)d_in[4];
    const float* W_node         = (const float*)d_in[5];
    const float* W_rad_e        = (const float*)d_in[6];
    const float* W_edge         = (const float*)d_in[7];
    const float* W_node_lin     = (const float*)d_in[8];
    const float* b_node_lin     = (const float*)d_in[9];
    const float* W_edge_lin     = (const float*)d_in[10];
    const float* b_edge_lin     = (const float*)d_in[11];

    float* ws = (float*)d_ws;
    float* dist_emb = ws; ws += (size_t)N_EDGES * NG_DIM;
    float* Rm       = ws; ws += (size_t)N_EDGES * 9;
    float* node_emb = ws; ws += (size_t)N_NODES * D_DIM;
    float* edge_emb = ws; ws += (size_t)N_EDGES * D_DIM;
    float* rad      = ws; ws += (size_t)N_EDGES * NG_DIM;
    float* msg      = ws; ws += (size_t)N_EDGES * D_DIM;   // reused as e / X_edge
    float* agg      = ws; ws += (size_t)N_NODES * D_DIM;   // reused as X_node

    const int* src = edge_index;
    const int* tgt = edge_index + N_EDGES;

    auto gemm = [&](const float* A, const float* B, const float* bias, float* Out,
                    int M, int N, int K, int flags) {
        dim3 grid((M + BM - 1) / BM, N / BN);
        k_gemm_wmma<<<grid, 256, 0, stream>>>(A, B, bias, Out, M, N, K, flags);
    };

    k_smear<<<N_EDGES, NG_DIM, 0, stream>>>(edge_attr, dist_emb);
    k_rotmat<<<(N_EDGES + 255) / 256, 256, 0, stream>>>(edge_attr, Rm);
    k_init_node<<<(N_NODES * D_DIM + 255) / 256, 256, 0, stream>>>(
        atomic_numbers, sphere_emb, node_emb);
    k_init_edge<<<(N_EDGES * D_DIM + 255) / 256, 256, 0, stream>>>(
        dist_emb, edge_emb);

    for (int i = 0; i < 2; ++i) {
        // rad_n = silu(dist_emb @ W_rad_n[i])
        gemm(dist_emb, W_rad_n + (size_t)i * NG_DIM * C_DIM, nullptr, rad,
             N_EDGES, C_DIM, NG_DIM, /*SILU*/2);
        // m = rotate(silu(node[src]+node[tgt]+edge) * rad)
        k_message<<<N_EDGES, 256, 0, stream>>>(node_emb, edge_emb, rad, src, tgt,
                                               Rm, msg, /*transposeR=*/0);
        // agg = segment_sum(m, tgt)
        k_zero<<<(N_NODES * D_DIM + 255) / 256, 256, 0, stream>>>(
            agg, (unsigned)(N_NODES * D_DIM));
        k_scatter<<<(N_EDGES * D_DIM + 255) / 256, 256, 0, stream>>>(msg, tgt, agg);
        // node_emb += agg @ W_node[i]
        gemm(agg, W_node + (size_t)i * C_DIM * C_DIM, nullptr, node_emb,
             N_NODES * NCOEF, C_DIM, C_DIM, /*ACCUM*/1);
        // rad_e = silu(dist_emb @ W_rad_e[i])
        gemm(dist_emb, W_rad_e + (size_t)i * NG_DIM * C_DIM, nullptr, rad,
             N_EDGES, C_DIM, NG_DIM, /*SILU*/2);
        // e = rotate^T(silu(node[src]+node[tgt]+edge) * rad)
        k_message<<<N_EDGES, 256, 0, stream>>>(node_emb, edge_emb, rad, src, tgt,
                                               Rm, msg, /*transposeR=*/1);
        // edge_emb += e @ W_edge[i]
        gemm(msg, W_edge + (size_t)i * C_DIM * C_DIM, nullptr, edge_emb,
             N_EDGES * NCOEF, C_DIM, C_DIM, /*ACCUM*/1);
    }

    float* node_out = (float*)d_out;
    float* edge_out = node_out + (size_t)N_NODES * D_DIM;

    // node_out = irreps(node_emb) @ W_node_lin + b_node_lin
    k_irreps<<<(N_NODES * D_DIM + 255) / 256, 256, 0, stream>>>(node_emb, agg, N_NODES);
    gemm(agg, W_node_lin, b_node_lin, node_out, N_NODES, D_DIM, D_DIM, /*BIAS*/4);

    // edge_out = irreps(edge_emb) @ W_edge_lin + b_edge_lin  (dominant GEMM)
    k_irreps<<<(N_EDGES * D_DIM + 255) / 256, 256, 0, stream>>>(edge_emb, msg, N_EDGES);
    gemm(msg, W_edge_lin, b_edge_lin, edge_out, N_EDGES, D_DIM, D_DIM, /*BIAS*/4);
}
